// VectorQuantizerEMA1D_52544629899302
// MI455X (gfx1250) — matrix operand order repaired and checked by reading, not compiled, not run on hardware
//
#include <hip/hip_runtime.h>

typedef __attribute__((ext_vector_type(16))) __bf16 v16bf;
typedef __attribute__((ext_vector_type(8)))  __bf16 v8bf;
typedef __attribute__((ext_vector_type(8)))  float  v8f;
typedef __attribute__((ext_vector_type(4)))  unsigned int v4u;
typedef __attribute__((ext_vector_type(8)))  int v8i;
typedef __attribute__((ext_vector_type(4)))  int v4i;

#define B_   4
#define D_   512
#define TQ_  4096
#define NROW 16384   // B_*TQ_
#define KCB  8192    // codebook entries
#define NEG_INF (-3.402823466e38f)
#define GB    32     // codes per staged B tile
#define BPAD  520    // LDS row stride (elements) after TDM padding: 1024B + 16B pad
#define MPB   128    // rows per block (8 waves x 16)

// ---------------------------------------------------------------------------
// Normalize codebook rows -> bf16.  One wave per row (8 rows / 256-thr block).
// ---------------------------------------------------------------------------
__global__ void k_norm_emb(const float* __restrict__ emb, __bf16* __restrict__ eb) {
    int tid  = threadIdx.x;
    int wave = tid >> 5;
    int lane = tid & 31;
    int row  = blockIdx.x * 8 + wave;
    const float* ep = emb + (size_t)row * D_;
    float p = 0.f;
#pragma unroll
    for (int j = 0; j < 16; ++j) { float v = ep[j * 32 + lane]; p += v * v; }
#pragma unroll
    for (int off = 16; off > 0; off >>= 1) p += __shfl_xor(p, off, 32);
    float scale = 1.0f / fmaxf(sqrtf(p), 1e-12f);
    __bf16* op = eb + (size_t)row * D_;
#pragma unroll
    for (int j = 0; j < 16; ++j) op[j * 32 + lane] = (__bf16)(ep[j * 32 + lane] * scale);
}

// ---------------------------------------------------------------------------
// Normalize z columns ([b,d,t] -> flat rows [n][k], bf16, row-major).
// ---------------------------------------------------------------------------
__global__ void k_norm_z(const float* __restrict__ z, __bf16* __restrict__ ez) {
    __shared__ float  s_part[4][64];
    __shared__ float  s_scale[64];
    __shared__ __bf16 s_tile[64][72];

    int tid = threadIdx.x;
    int gt  = blockIdx.x * 64;
    int b   = gt >> 12;
    int tz  = gt & (TQ_ - 1);
    int q   = tid >> 6;
    int tl  = tid & 63;
    const float* zb = z + (size_t)b * D_ * TQ_ + tz;

    float p = 0.f;
    for (int k = q; k < D_; k += 4) { float v = zb[(size_t)k * TQ_ + tl]; p += v * v; }
    s_part[q][tl] = p;
    __syncthreads();
    if (tid < 64) {
        float s = s_part[0][tid] + s_part[1][tid] + s_part[2][tid] + s_part[3][tid];
        s_scale[tid] = 1.0f / fmaxf(sqrtf(s), 1e-12f);
    }
    __syncthreads();

    for (int kc = 0; kc < D_; kc += 64) {
#pragma unroll
        for (int i = 0; i < 16; ++i) {
            int kl = (tid >> 6) + i * 4;
            float v = zb[(size_t)(kc + kl) * TQ_ + tl] * s_scale[tl];
            s_tile[tl][kl] = (__bf16)v;
        }
        __syncthreads();
        int tr = tid >> 2, c0 = (tid & 3) * 16;
        v16bf out;
#pragma unroll
        for (int j = 0; j < 16; ++j) out[j] = s_tile[tr][c0 + j];
        *(v16bf*)(ez + (size_t)(gt + tr) * D_ + kc + c0) = out;
        __syncthreads();
    }
}

// ---------------------------------------------------------------------------
// Fused GEMM + row argmax with TDM-staged B tiles.
//   Block = 8 waves; wave w owns rows m0+w*16..+15 (A tile in registers).
//   All waves share double-buffered LDS B tiles of GB=32 codes, staged by the
//   Tensor Data Mover (pad_enable gives LDS row stride 1040B -> conflict-free).
//   Each wave scans ALL codes -> argmax is wave-local.
// ---------------------------------------------------------------------------
__global__ void __launch_bounds__(256)
k_vq_argmax(const __bf16* __restrict__ ez,
            const __bf16* __restrict__ eb,
            int* __restrict__ idx) {
    __shared__ __bf16 s_b[2][GB * BPAD];   // 2 x 33,280 B

    int tid  = threadIdx.x;
    int wave = tid >> 5;
    int lane = tid & 31;
    int m0   = blockIdx.x * MPB + wave * 16;

    // --- A fragments (contiguous k order, k = 16*(lane/16) + s) in registers, loaded once.
    // lane<16: M=lane, K{0-15}; lane>=16: M=lane-16, K{16-31}.
    int rA  = lane & 15;
    int k0A = (lane & 16) ? 16 : 0;
    const __bf16* aptr = ez + (size_t)(m0 + rA) * D_ + k0A;
    v16bf afrag[16];
#pragma unroll
    for (int kk = 0; kk < 16; ++kk) {
        v8bf lo = *(const v8bf*)(aptr + kk * 32);
        v8bf hi = *(const v8bf*)(aptr + kk * 32 + 8);
        v16bf a;
#pragma unroll
        for (int j = 0; j < 8; ++j) { a[j] = lo[j]; a[j + 8] = hi[j]; }
        afrag[kk] = a;
    }

    // --- TDM descriptor group 1 (constant): 2D tensor, data_size=8B units.
    // tensor_dim0 = tile_dim0 = 128 units (1024B row), tile_dim1 = GB rows,
    // tensor_dim0_stride = 128, pad: 16B after every 256 DWORDs (1024B).
    v8i g1;
    g1[0] = (int)((3u << 16) | (1u << 20) | (7u << 22) | (3u << 25));
    g1[1] = (int)(128u << 16);    // tensor_dim0[15:0]
    g1[2] = (int)(8192u << 16);   // tensor_dim0 hi = 0, tensor_dim1[15:0] = 8192
    g1[3] = (int)(128u << 16);    // tensor_dim1 hi = 0, tile_dim0 = 128
    g1[4] = GB;                   // tile_dim1 = 32, tile_dim2 = 0
    g1[5] = 128;                  // tensor_dim0_stride lo
    g1[6] = 0;
    g1[7] = 0;
    v4i gz  = {0, 0, 0, 0};
    v8i gz8 = {0, 0, 0, 0, 0, 0, 0, 0};

    unsigned lds_base0 = (unsigned)(uintptr_t)(&s_b[0][0]);
    unsigned lds_base1 = (unsigned)(uintptr_t)(&s_b[1][0]);

    // B-fragment addressing: lane%16 = code column within 16-tile;
    // lanes 0-15 hold K 0..15, lanes 16-31 hold K 16..31.
    int colB = lane & 15;
    int kbB  = (lane & 16) ? 16 : 0;

    float bestv[8];
    int   besti[8];
#pragma unroll
    for (int r = 0; r < 8; ++r) { bestv[r] = NEG_INF; besti[r] = 0; }

    const int nGroups = KCB / GB;   // 256

    if (wave == 0) {
        v4u g0;
        g0[0] = 1u;
        g0[1] = lds_base0;
        unsigned long long ga = (unsigned long long)(uintptr_t)eb;
        g0[2] = (unsigned)(ga & 0xffffffffu);
        g0[3] = (unsigned)((ga >> 32) & 0x01ffffffu) | (2u << 30);
        __builtin_amdgcn_tensor_load_to_lds(g0, g1, gz, gz, gz8, 0);
    }

    for (int g = 0; g < nGroups; ++g) {
        int buf = g & 1;
        if (wave == 0) __builtin_amdgcn_s_wait_tensorcnt(0);
        __syncthreads();   // tile g visible to all; prior reads of buf^1 done
        if (wave == 0 && g + 1 < nGroups) {
            v4u g0;
            g0[0] = 1u;
            g0[1] = buf ? lds_base0 : lds_base1;
            unsigned long long ga = (unsigned long long)(uintptr_t)eb
                                  + (unsigned long long)(g + 1) * (GB * 1024u);
            g0[2] = (unsigned)(ga & 0xffffffffu);
            g0[3] = (unsigned)((ga >> 32) & 0x01ffffffu) | (2u << 30);
            __builtin_amdgcn_tensor_load_to_lds(g0, g1, gz, gz, gz8, 0);
        }

        const __bf16* bt = &s_b[buf][0];
        const __bf16* p0 = bt + (colB)      * BPAD + kbB;   // tile 0: codes n0..n0+15
        const __bf16* p1 = bt + (16 + colB) * BPAD + kbB;   // tile 1: codes n0+16..n0+31
        v8f acc0 = {}, acc1 = {};
#pragma unroll
        for (int kk = 0; kk < 16; ++kk) {
            v8bf l0 = *(const v8bf*)(p0 + kk * 32);
            v8bf h0 = *(const v8bf*)(p0 + kk * 32 + 8);
            v8bf l1 = *(const v8bf*)(p1 + kk * 32);
            v8bf h1 = *(const v8bf*)(p1 + kk * 32 + 8);
            v16bf f0, f1;
#pragma unroll
            for (int j = 0; j < 8; ++j) {
                f0[j] = l0[j]; f0[j + 8] = h0[j];
                f1[j] = l1[j]; f1[j + 8] = h1[j];
            }
            acc0 = __builtin_amdgcn_wmma_f32_16x16x32_bf16(false, afrag[kk], false, f0, (short)0, acc0, false, false);
            acc1 = __builtin_amdgcn_wmma_f32_16x16x32_bf16(false, afrag[kk], false, f1, (short)0, acc1, false, false);
        }
        int n = g * GB + colB;
#pragma unroll
        for (int r = 0; r < 8; ++r) {
            if (acc0[r] > bestv[r]) { bestv[r] = acc0[r]; besti[r] = n;      }
            if (acc1[r] > bestv[r]) { bestv[r] = acc1[r]; besti[r] = n + 16; }
        }
    }

    // C layout: VGPR r = row r (lanes 0-15) / row r+8 (lanes 16-31), col=lane%16.
    // Argmax-reduce across the 16-lane halves (ties -> min idx).
#pragma unroll
    for (int r = 0; r < 8; ++r) {
#pragma unroll
        for (int off = 1; off < 16; off <<= 1) {
            float ov = __shfl_xor(bestv[r], off, 32);
            int   oi = __shfl_xor(besti[r], off, 32);
            if (ov > bestv[r] || (ov == bestv[r] && oi < besti[r])) {
                bestv[r] = ov; besti[r] = oi;
            }
        }
    }
    if ((lane & 15) == 0) {
        int half = (lane >> 4) * 8;   // lane0 -> rows 0-7, lane16 -> rows 8-15
#pragma unroll
        for (int r = 0; r < 8; ++r) idx[m0 + half + r] = besti[r];
    }
}

// ---------------------------------------------------------------------------
// Gather + outputs.  Coalesced over t; idx written as float.
// ---------------------------------------------------------------------------
__global__ void k_gather(const float* __restrict__ z,
                         const float* __restrict__ emb,
                         const int* __restrict__ idx,
                         float* __restrict__ out0,
                         float* __restrict__ out1,
                         float* __restrict__ out2) {
    int n = blockIdx.x * 256 + threadIdx.x;
    int b = n >> 12, t = n & (TQ_ - 1);
    int id = idx[n];
    out1[n] = (float)id;
    const float* ep = emb + (size_t)id * D_;
    const float* zp = z   + (size_t)b * D_ * TQ_ + t;
    float* o0 = out0 + (size_t)b * D_ * TQ_ + t;
    float* o2 = out2 + (size_t)b * D_ * TQ_ + t;
    for (int k = 0; k < D_; ++k) {
        float qv = ep[k];
        float zz = zp[(size_t)k * TQ_];
        o0[(size_t)k * TQ_] = zz + (qv - zz);
        o2[(size_t)k * TQ_] = qv;
    }
}

// ---------------------------------------------------------------------------
extern "C" void kernel_launch(void* const* d_in, const int* in_sizes, int n_in,
                              void* d_out, int out_size, void* d_ws, size_t ws_size,
                              hipStream_t stream) {
    const float* z   = (const float*)d_in[0];   // [4,512,4096]
    const float* emb = (const float*)d_in[1];   // [8192,512]

    char* ws = (char*)d_ws;
    __bf16* ez = (__bf16*)ws;                                   // 16 MB
    __bf16* eb = (__bf16*)(ws + (size_t)NROW * D_ * 2);         //  8 MB
    int*   idx = (int*)(ws + (size_t)NROW * D_ * 2 + (size_t)KCB * D_ * 2);

    float* out0 = (float*)d_out;                 // z_q_st : 8388608
    float* out1 = out0 + (size_t)B_ * D_ * TQ_;  // idx    : 16384
    float* out2 = out1 + (size_t)B_ * TQ_;       // z_q    : 8388608

    k_norm_emb <<<KCB / 8,    256, 0, stream>>>(emb, eb);
    k_norm_z   <<<NROW / 64,  256, 0, stream>>>(z, ez);
    k_vq_argmax<<<NROW / MPB, 256, 0, stream>>>(ez, eb, idx);
    k_gather   <<<NROW / 256, 256, 0, stream>>>(z, emb, idx, out0, out1, out2);
}
